// RNN_13975823581859
// MI455X (gfx1250) — matrix operand order repaired and checked by reading, not compiled, not run on hardware
//
#include <hip/hip_runtime.h>
#include <hip/hip_bf16.h>
#include <stdint.h>

#define DEV_INLINE __device__ __forceinline__

// ---------------------------------------------------------------------------
// Problem dimensions (fixed by the reference)
// ---------------------------------------------------------------------------
constexpr int T  = 224;
constexpr int I  = 672;          // 21 * 32
constexpr int H  = 256;
constexpr int B  = 256;
constexpr int TI = T * I;        // 150528 elements per batch row of x

typedef __attribute__((ext_vector_type(16))) __bf16 v16bf;
typedef __attribute__((ext_vector_type(8)))  float  v8f;

union FragBF {
    v16bf    v;
    uint32_t u[8];
    uint4    q[2];
};

// round-to-nearest-even f32 -> bf16
DEV_INLINE uint32_t pack_bf16(float a, float b) {
    uint32_t ua = __builtin_bit_cast(uint32_t, a);
    uint32_t ub = __builtin_bit_cast(uint32_t, b);
    ua += 0x7FFFu + ((ua >> 16) & 1u);
    ub += 0x7FFFu + ((ub >> 16) & 1u);
    return (ua >> 16) | (ub & 0xFFFF0000u);
}
DEV_INLINE unsigned short f32_to_bf16(float a) {
    uint32_t ua = __builtin_bit_cast(uint32_t, a);
    ua += 0x7FFFu + ((ua >> 16) & 1u);
    return (unsigned short)(ua >> 16);
}
DEV_INLINE float bf16_to_f32(unsigned short s) {
    uint32_t u = ((uint32_t)s) << 16;
    return __builtin_bit_cast(float, u);
}

// ---------------------------------------------------------------------------
// Kernel 0: bulk f32 -> packed bf16 conversion (memory bound; 8 elems/thread).
// count must be a multiple of 8 (all our sizes are).
// ---------------------------------------------------------------------------
__global__ __launch_bounds__(256) void cvt_bf16_kernel(
    const float* __restrict__ src, uint32_t* __restrict__ dst, int n8)
{
    const int i = blockIdx.x * 256 + threadIdx.x;
    if (i >= n8) return;
    const float4* p = reinterpret_cast<const float4*>(src) + (size_t)i * 2;
    float4 a = p[0], b = p[1];
    uint4 o;
    o.x = pack_bf16(a.x, a.y); o.y = pack_bf16(a.z, a.w);
    o.z = pack_bf16(b.x, b.y); o.w = pack_bf16(b.z, b.w);
    reinterpret_cast<uint4*>(dst)[i] = o;
}

// ---------------------------------------------------------------------------
// Fragment loaders from pre-packed bf16 memory (pure b128 loads, no VALU).
// A-operand (16x32): lane holds row m=L&15, K = lo+{0..7} in u0..u3 and
// lo+16+{0..7} in u4..u7, lo = 8*(L>=16).  `e` = element offset of (row, k0+lo).
DEV_INLINE void load_afrag16(FragBF& f, const unsigned short* __restrict__ base, size_t e) {
    const uint4* p = reinterpret_cast<const uint4*>(base + e);
    f.q[0] = p[0];       // K = lo .. lo+7
    f.q[1] = p[2];       // K = lo+16 .. lo+23   (+16 elems = +32 bytes = +2 uint4)
}
// B-operand (32x16, B = W^T): lane holds column n=L&15,
// K = lo+{0..7} in u0..u3 and lo+16+{0..7} in u4..u7, lo = 8*(L>=16).  `e` = offset of W[n][kb+lo].
DEV_INLINE void load_bfrag16(FragBF& f, const unsigned short* __restrict__ base, size_t e) {
    const uint4* p = reinterpret_cast<const uint4*>(base + e);
    f.q[0] = p[0];
    f.q[1] = p[2];
}

// ---------------------------------------------------------------------------
// Kernel 1: xp[t*B + b, h] = sum_i x[b, t, i] * W_ih0[h, i] + b_ih0[h]
// One wave computes a 16(M) x 64(N) strip; K loop of 21 x WMMA 16x16x32 bf16.
// Inner loop: 2 + 8 b128 loads + 4 WMMA, zero conversion VALU.
// ---------------------------------------------------------------------------
__global__ __launch_bounds__(256) void xproj_kernel(
    const unsigned short* __restrict__ xbf,
    const unsigned short* __restrict__ Wih0bf,
    const float* __restrict__ bih0, float* __restrict__ xp)
{
    const int lane  = threadIdx.x & 31;
    const int wid   = threadIdx.x >> 5;
    const int task  = blockIdx.x * 8 + wid;      // 0..14335
    const int mtile = task >> 2;                 // 0..3583
    const int ng    = task & 3;                  // N-group (4 tiles of 16)
    const int t     = mtile >> 4;
    const int b0    = (mtile & 15) << 4;
    const int l15   = lane & 15;
    const int hi    = lane >> 4;

    const size_t arow = (size_t)(b0 + l15) * TI + (size_t)t * I;

    v8f acc[4];
    #pragma unroll
    for (int j = 0; j < 4; ++j) {
        const float bj = bih0[ng * 64 + j * 16 + l15];
        #pragma unroll
        for (int v = 0; v < 8; ++v) acc[j][v] = bj;
    }

    #pragma unroll 1
    for (int k0 = 0; k0 < I; k0 += 32) {
        FragBF a;
        load_afrag16(a, xbf, arow + k0 + hi * 8);
        #pragma unroll
        for (int j = 0; j < 4; ++j) {
            const int n = ng * 64 + j * 16 + l15;
            FragBF b;
            load_bfrag16(b, Wih0bf, (size_t)n * I + k0 + hi * 8);
            acc[j] = __builtin_amdgcn_wmma_f32_16x16x32_bf16(
                false, a.v, false, b.v, (short)0, acc[j], false, false);
        }
    }

    #pragma unroll
    for (int j = 0; j < 4; ++j) {
        const int n = ng * 64 + j * 16 + l15;
        #pragma unroll
        for (int v = 0; v < 8; ++v) {
            const int m = v + hi * 8;
            xp[((size_t)t * B + b0 + m) * H + n] = acc[j][v];
        }
    }
}

// ---------------------------------------------------------------------------
// Kernel 2: serial scan.  16 independent batch chunks (one WGP each, no
// cross-WG sync).  16 waves/WG; wave w owns output columns [16w,16w+16) and
// keeps all 24 weight B-fragments register resident across the 224-step loop.
// h0/h1: double-buffered bf16 in LDS, row stride 264 halfwords
// (132 dwords == 4 mod 64 banks -> distinct banks for the 16 rows).
// ---------------------------------------------------------------------------
__global__ __launch_bounds__(512) void rnn_scan_kernel(
    const float* __restrict__ xp,
    const unsigned short* __restrict__ Whh0bf, const float* __restrict__ bhh0,
    const unsigned short* __restrict__ Wih1bf, const float* __restrict__ bih1,
    const unsigned short* __restrict__ Whh1bf, const float* __restrict__ bhh1,
    const float* __restrict__ Wfc,  const float* __restrict__ bfc,
    float* __restrict__ out)
{
    __shared__ __align__(16) unsigned short h0b[2][16][264];
    __shared__ __align__(16) unsigned short h1b[2][16][264];

    const int lane = threadIdx.x & 31;
    const int w    = threadIdx.x >> 5;     // 0..15
    const int l15  = lane & 15;
    const int hi   = lane >> 4;
    const int bch  = blockIdx.x << 4;      // batch chunk start
    const int n    = (w << 4) + l15;       // output column of this lane
    const int m    = l15;                  // A-operand row of this lane

    // Preload all weight fragments (loop-invariant, register resident).
    FragBF wf0[8], wf1[8], wf2[8];
    #pragma unroll
    for (int j = 0; j < 8; ++j) {
        const size_t off = (size_t)n * H + j * 32 + hi * 8;
        load_bfrag16(wf0[j], Whh0bf, off);
        load_bfrag16(wf1[j], Wih1bf, off);
        load_bfrag16(wf2[j], Whh1bf, off);
    }
    const float bias0 = bhh0[n];
    const float bias1 = bih1[n] + bhh1[n];

    // Zero the initial hidden state (buffer 0 is read at t=0).
    for (int i = threadIdx.x; i < 2 * 16 * 264; i += 512) {
        ((unsigned short*)h0b)[i] = 0;
        ((unsigned short*)h1b)[i] = 0;
    }
    __syncthreads();

    #pragma unroll 1
    for (int t = 0; t < T; ++t) {
        const int pr = t & 1, pw = pr ^ 1;

        // ---- layer 0: h0 = relu(xp_t + h0 @ Whh0^T + bhh0) ----
        v8f acc;
        #pragma unroll
        for (int v = 0; v < 8; ++v)
            acc[v] = xp[((size_t)t * B + bch + v + hi * 8) * H + n];
        #pragma unroll
        for (int j = 0; j < 8; ++j) {
            FragBF a;
            const uint4* prow = reinterpret_cast<const uint4*>(&h0b[pr][m][0]);
            a.q[0] = prow[j * 4 + hi];
            a.q[1] = prow[j * 4 + hi + 2];
            acc = __builtin_amdgcn_wmma_f32_16x16x32_bf16(
                false, a.v, false, wf0[j].v, (short)0, acc, false, false);
        }
        #pragma unroll
        for (int v = 0; v < 8; ++v) {
            const float r = fmaxf(acc[v] + bias0, 0.0f);
            h0b[pw][v + hi * 8][n] = f32_to_bf16(r);
        }
        __syncthreads();

        // ---- layer 1: h1 = relu(h0_new @ Wih1^T + h1 @ Whh1^T + biases) ----
        v8f acc1;
        #pragma unroll
        for (int v = 0; v < 8; ++v) acc1[v] = bias1;
        #pragma unroll
        for (int j = 0; j < 8; ++j) {
            FragBF a;
            const uint4* prow = reinterpret_cast<const uint4*>(&h0b[pw][m][0]);
            a.q[0] = prow[j * 4 + hi];
            a.q[1] = prow[j * 4 + hi + 2];
            acc1 = __builtin_amdgcn_wmma_f32_16x16x32_bf16(
                false, a.v, false, wf1[j].v, (short)0, acc1, false, false);
        }
        #pragma unroll
        for (int j = 0; j < 8; ++j) {
            FragBF a;
            const uint4* prow = reinterpret_cast<const uint4*>(&h1b[pr][m][0]);
            a.q[0] = prow[j * 4 + hi];
            a.q[1] = prow[j * 4 + hi + 2];
            acc1 = __builtin_amdgcn_wmma_f32_16x16x32_bf16(
                false, a.v, false, wf2[j].v, (short)0, acc1, false, false);
        }
        #pragma unroll
        for (int v = 0; v < 8; ++v) {
            const float r = fmaxf(acc1[v], 0.0f);
            h1b[pw][v + hi * 8][n] = f32_to_bf16(r);
        }
        __syncthreads();
    }

    // t = 223 wrote parity 0: final h1 is in h1b[0].  FC: out = h1 @ Wfc^T + bfc.
    if (threadIdx.x < 32) {
        const int b_idx = l15;
        const int o     = hi;            // O = 2
        float s = bfc[o];
        for (int h = 0; h < H; ++h)
            s += bf16_to_f32(h1b[0][b_idx][h]) * Wfc[o * H + h];
        out[(bch + b_idx) * 2 + o] = s;
    }
}

// ---------------------------------------------------------------------------
extern "C" void kernel_launch(void* const* d_in, const int* in_sizes, int n_in,
                              void* d_out, int out_size, void* d_ws, size_t ws_size,
                              hipStream_t stream) {
    const float* x    = (const float*)d_in[0];
    const float* Wih0 = (const float*)d_in[1];
    const float* Whh0 = (const float*)d_in[2];
    const float* bih0 = (const float*)d_in[3];
    const float* bhh0 = (const float*)d_in[4];
    const float* Wih1 = (const float*)d_in[5];
    const float* Whh1 = (const float*)d_in[6];
    const float* bih1 = (const float*)d_in[7];
    const float* bhh1 = (const float*)d_in[8];
    const float* Wfc  = (const float*)d_in[9];
    const float* bfc  = (const float*)d_in[10];
    float* out = (float*)d_out;

    // ---- workspace layout (all 256B aligned) ----
    char* ws = (char*)d_ws;
    float*          xp      = (float*)ws;                       // 58,720,256 B
    size_t off = (size_t)T * B * H * sizeof(float);
    unsigned short* xbf     = (unsigned short*)(ws + off);      // 77,070,336 B
    off += (size_t)B * TI * sizeof(unsigned short);
    unsigned short* Wih0bf  = (unsigned short*)(ws + off);      // 344,064 B
    off += (size_t)H * I * sizeof(unsigned short);
    unsigned short* Whh0bf  = (unsigned short*)(ws + off);      // 131,072 B
    off += (size_t)H * H * sizeof(unsigned short);
    unsigned short* Wih1bf  = (unsigned short*)(ws + off);
    off += (size_t)H * H * sizeof(unsigned short);
    unsigned short* Whh1bf  = (unsigned short*)(ws + off);

    // Phase 0: one-time f32 -> bf16 conversions (memory bound, ~10 us total).
    const int nx8 = (B * TI) / 8;                 // 4,816,896
    cvt_bf16_kernel<<<(nx8 + 255) / 256, 256, 0, stream>>>(x, (uint32_t*)xbf, nx8);
    const int nw8 = (H * I) / 8;                  // 21,504
    cvt_bf16_kernel<<<(nw8 + 255) / 256, 256, 0, stream>>>(Wih0, (uint32_t*)Wih0bf, nw8);
    const int nh8 = (H * H) / 8;                  // 8,192
    cvt_bf16_kernel<<<(nh8 + 255) / 256, 256, 0, stream>>>(Whh0, (uint32_t*)Whh0bf, nh8);
    cvt_bf16_kernel<<<(nh8 + 255) / 256, 256, 0, stream>>>(Wih1, (uint32_t*)Wih1bf, nh8);
    cvt_bf16_kernel<<<(nh8 + 255) / 256, 256, 0, stream>>>(Whh1, (uint32_t*)Whh1bf, nh8);

    // Phase 1: big parallel GEMM (input projection), 14336 wave-tiles.
    xproj_kernel<<<1792, 256, 0, stream>>>(xbf, Wih0bf, bih0, xp);

    // Phase 2: 16 independent batch-chunk scans, one WGP each.
    rnn_scan_kernel<<<16, 512, 0, stream>>>(xp, Whh0bf, bhh0, Wih1bf, bih1,
                                            Whh1bf, bhh1, Wfc, bfc, out);
}
